// FasterRCNN_31275951850291
// MI455X (gfx1250) — matrix operand order repaired and checked by reading, not compiled, not run on hardware
//
#include <hip/hip_runtime.h>
#include <math.h>

// ---------- CDNA5 WMMA vector types (POD, union-safe) ----------
typedef __attribute__((ext_vector_type(16))) _Float16 v16h;
typedef __attribute__((ext_vector_type(8)))  float    v8f;
typedef __attribute__((ext_vector_type(4)))  unsigned int u32x4;

union H16 { u32x4 u[2]; v16h v; _Float16 h[16]; };
union F8  { v8f v; float f[8]; };

#define NROI   3000
#define MPAD   3008          // 188 tiles of 16
#define MT     188
#define KFEAT  12544         // 256*7*7
#define DHID   1024
#define NCLS   20

// =====================================================================
// 1) Pack fp32 weight (K x N row-major) into fp16 WMMA B-fragment layout:
//    Bp[((kt*Ntiles + nt)*32 + lane)*16 + e], lane = (kin>>4)*16 + nin,
//    e = kin & 15   (lanes 0-15: K 0..15 of tile, lanes 16-31: K 16..31)
// =====================================================================
__global__ void pack_w_kernel(const float* __restrict__ w, _Float16* __restrict__ bp,
                              int K, int N, int Npad) {
  long gid = (long)blockIdx.x * blockDim.x + threadIdx.x;
  long total = (long)K * Npad;
  if (gid >= total) return;
  int k = (int)(gid / Npad);
  int n = (int)(gid % Npad);
  float v = (n < N) ? w[(size_t)k * N + n] : 0.0f;
  int kt = k >> 5, kin = k & 31;
  int nt = n >> 4, nin = n & 15;
  int lane = ((kin >> 4) << 4) | nin;
  int e = kin & 15;
  int Ntiles = Npad >> 4;
  bp[((((size_t)kt * Ntiles + nt) * 32 + lane) << 4) + e] = (_Float16)v;
}

// =====================================================================
// 2) ROI-Align: block = one ROI (3008 blocks; >=3000 writes zero pad rows)
//    thread = channel (256). Output A row-major fp16: A[m][c*49 + iy*7 + ix]
// =====================================================================
__global__ void roi_align_kernel(const float* __restrict__ p3,
                                 const float* __restrict__ p4,
                                 const float* __restrict__ p5,
                                 const float* __restrict__ pr3,
                                 const float* __restrict__ pr4,
                                 const float* __restrict__ pr5,
                                 _Float16* __restrict__ A) {
  int m = blockIdx.x;
  int t = threadIdx.x;
  size_t rowbase = (size_t)m * KFEAT;
  if (m >= NROI) {
    for (int s = 0; s < 49; ++s) A[rowbase + (size_t)t * 49 + s] = (_Float16)0.0f;
    return;
  }
  __shared__ int   sx0[49], sx1[49], sy0[49], sy1[49];
  __shared__ float swx[49], swy[49];

  int level = m / 1000, pi = m % 1000;
  const float* feat;
  const float* props;
  int H;
  float stride;
  if (level == 0)      { feat = p3; props = pr3; H = 128; stride = 8.0f;  }
  else if (level == 1) { feat = p4; props = pr4; H = 64;  stride = 16.0f; }
  else                 { feat = p5; props = pr5; H = 32;  stride = 32.0f; }
  int W = H;

  if (t < 49) {
    float x1 = props[pi * 4 + 0] / stride;
    float y1 = props[pi * 4 + 1] / stride;
    float x2 = props[pi * 4 + 2] / stride;
    float y2 = props[pi * 4 + 3] / stride;
    int iy = t / 7, ix = t % 7;
    float bx = (ix + 0.5f) / 7.0f;
    float by = (iy + 0.5f) / 7.0f;
    float xs = x1 + bx * (x2 - x1) - 0.5f;
    float ys = y1 + by * (y2 - y1) - 0.5f;
    float x0 = floorf(xs), y0 = floorf(ys);
    swx[t] = xs - x0;
    swy[t] = ys - y0;
    int x0i = min(max((int)x0, 0), W - 1);
    int y0i = min(max((int)y0, 0), H - 1);
    sx0[t] = x0i; sx1[t] = min(x0i + 1, W - 1);
    sy0[t] = y0i; sy1[t] = min(y0i + 1, H - 1);
  }
  __syncthreads();

  const float* fc = feat + (size_t)t * H * W;
  size_t ob = rowbase + (size_t)t * 49;
  for (int s = 0; s < 49; ++s) {
    int r0 = sy0[s] * W, r1 = sy1[s] * W;
    float v00 = fc[r0 + sx0[s]];
    float v01 = fc[r0 + sx1[s]];
    float v10 = fc[r1 + sx0[s]];
    float v11 = fc[r1 + sx1[s]];
    float wx1 = swx[s], wy1 = swy[s];
    float o = (1.0f - wy1) * ((1.0f - wx1) * v00 + wx1 * v01)
            +         wy1  * ((1.0f - wx1) * v10 + wx1 * v11);
    A[ob + s] = (_Float16)o;
  }
}

// =====================================================================
// 3) WMMA GEMM: out[MPAD x N] = relu(A[MPAD x K] * Bp + bias)
//    One wave per 16x64 output strip (1 m-tile x 4 n-tiles): the A
//    fragment is loaded once per k-step and reused by 4 back-to-back
//    v_wmma_f32_16x16x32_f16. B fragments for the 4 n-tiles are
//    contiguous in the packed layout (stride 512 halves) -> one clause.
//    Both operands are L2-resident (192 MB L2).
// =====================================================================
__global__ void gemm_wmma_kernel(const _Float16* __restrict__ A,
                                 const _Float16* __restrict__ Bp,
                                 const float* __restrict__ bias,
                                 _Float16* __restrict__ out,
                                 int Mtiles, int Ntiles, int K, int relu) {
  int wid    = (blockIdx.x * blockDim.x + threadIdx.x) >> 5;   // uniform per wave
  int nwaves = (gridDim.x * blockDim.x) >> 5;
  int lane   = threadIdx.x & 31;
  int lan15  = lane & 15;
  int ksel   = lane >> 4;        // 0: K 0..15 half, 1: K 16..31 half
  int KT     = K >> 5;
  int N      = Ntiles << 4;
  int NG     = Ntiles >> 2;      // groups of 4 n-tiles
  int tasks  = Mtiles * NG;

  for (int task = wid; task < tasks; task += nwaves) {
    int mt  = task / NG;
    int ng  = task - mt * NG;
    int nt0 = ng << 2;

    const _Float16* arow = A + (size_t)(mt * 16 + lan15) * K + ksel * 16;
    const _Float16* bcol = Bp + (((size_t)nt0 * 32 + lane) << 4);
    size_t bstep = ((size_t)Ntiles * 32) << 4;   // halves per k-tile of packed B

    F8 acc0, acc1, acc2, acc3;
#pragma unroll
    for (int e = 0; e < 8; ++e) {
      acc0.f[e] = 0.0f; acc1.f[e] = 0.0f; acc2.f[e] = 0.0f; acc3.f[e] = 0.0f;
    }

    for (int kt = 0; kt < KT; ++kt) {
      H16 af, bf0, bf1, bf2, bf3;
      const _Float16* ap = arow + kt * 32;
      af.u[0] = *(const u32x4*)(ap);        // K kbase..kbase+7
      af.u[1] = *(const u32x4*)(ap + 8);    // K kbase+8..kbase+15
      const _Float16* bpp = bcol + (size_t)kt * bstep;
      bf0.u[0] = *(const u32x4*)(bpp);        bf0.u[1] = *(const u32x4*)(bpp + 8);
      bf1.u[0] = *(const u32x4*)(bpp + 512);  bf1.u[1] = *(const u32x4*)(bpp + 520);
      bf2.u[0] = *(const u32x4*)(bpp + 1024); bf2.u[1] = *(const u32x4*)(bpp + 1032);
      bf3.u[0] = *(const u32x4*)(bpp + 1536); bf3.u[1] = *(const u32x4*)(bpp + 1544);
      acc0.v = __builtin_amdgcn_wmma_f32_16x16x32_f16(
          false, af.v, false, bf0.v, (short)0, acc0.v, false, false);
      acc1.v = __builtin_amdgcn_wmma_f32_16x16x32_f16(
          false, af.v, false, bf1.v, (short)0, acc1.v, false, false);
      acc2.v = __builtin_amdgcn_wmma_f32_16x16x32_f16(
          false, af.v, false, bf2.v, (short)0, acc2.v, false, false);
      acc3.v = __builtin_amdgcn_wmma_f32_16x16x32_f16(
          false, af.v, false, bf3.v, (short)0, acc3.v, false, false);
    }

    int mbase = mt * 16 + ksel * 8;
#pragma unroll
    for (int j = 0; j < 4; ++j) {
      const F8* accp = (j == 0) ? &acc0 : (j == 1) ? &acc1 : (j == 2) ? &acc2 : &acc3;
      int n = (nt0 + j) * 16 + lan15;
      float bv = bias[n];
#pragma unroll
      for (int e = 0; e < 8; ++e) {
        float v = accp->f[e] + bv;
        if (relu) v = fmaxf(v, 0.0f);
        out[(size_t)(mbase + e) * N + n] = (_Float16)v;
      }
    }
  }
}

// =====================================================================
// 4) Classifier: logits[m][c] = h2[m] . wcls[:,c] + bcls[c]   (fp32 acc)
// =====================================================================
__global__ void classifier_kernel(const _Float16* __restrict__ h2,
                                  const float* __restrict__ wcls,
                                  const float* __restrict__ bcls,
                                  float* __restrict__ logits) {
  int gid = blockIdx.x * blockDim.x + threadIdx.x;
  if (gid >= NROI * NCLS) return;
  int m = gid / NCLS, c = gid - m * NCLS;
  const _Float16* h = h2 + (size_t)m * DHID;
  float s = bcls[c];
  for (int k = 0; k < DHID; ++k) s += (float)h[k] * wcls[(size_t)k * NCLS + c];
  logits[gid] = s;
}

// =====================================================================
// 5) Scores: sigmoid/max/first-argmax; emit boxes/classes/scores to d_out
// =====================================================================
__global__ void score_kernel(const float* __restrict__ logits,
                             const float* __restrict__ pr3,
                             const float* __restrict__ pr4,
                             const float* __restrict__ pr5,
                             float* __restrict__ out,
                             float* __restrict__ scores,
                             int* __restrict__ classes) {
  int m = blockIdx.x * blockDim.x + threadIdx.x;
  if (m >= NROI) return;
  const float* lg = logits + m * NCLS;
  float best = lg[0];
  int bi = 0;
  for (int c = 1; c < NCLS; ++c) {
    float v = lg[c];
    if (v > best) { best = v; bi = c; }
  }
  float sc = 1.0f / (1.0f + expf(-best));
  scores[m] = sc;
  classes[m] = bi;
  int level = m / 1000, pi = m % 1000;
  const float* pp = (level == 0) ? pr3 : ((level == 1) ? pr4 : pr5);
  out[m * 4 + 0] = pp[pi * 4 + 0];
  out[m * 4 + 1] = pp[pi * 4 + 1];
  out[m * 4 + 2] = pp[pi * 4 + 2];
  out[m * 4 + 3] = pp[pi * 4 + 3];
  out[NROI * 4 + m]        = (float)bi;   // classes
  out[NROI * 4 + NROI + m] = sc;          // scores
}

// =====================================================================
// 6) NMS: single workgroup, 1024 threads. LDS bitonic sort (4096) +
//    class-offset boxes + exact sequential greedy suppression.
//    Dynamic LDS ~106 KB (WGP has 320 KB).
// =====================================================================
__global__ void nms_kernel(const float* __restrict__ pr3,
                           const float* __restrict__ pr4,
                           const float* __restrict__ pr5,
                           const float* __restrict__ scores,
                           const int* __restrict__ classes,
                           float* __restrict__ keep_out) {
  extern __shared__ char smem[];
  float* key  = (float*)smem;                 // 4096
  int*   idx  = (int*)(key + 4096);           // 4096
  float* sb   = (float*)(idx + 4096);         // 3000*4
  float* area = sb + NROI * 4;                // 3000
  int*   keep = (int*)(area + NROI);          // 3000
  float* red  = (float*)(keep + NROI);        // 1024
  int t = threadIdx.x;

  // ---- global max over all box coords ----
  float mx = -3.0e38f;
  for (int i = t; i < NROI * 4; i += 1024) {
    int b = i >> 2, j = i & 3;
    int lvl = b / 1000, pi = b % 1000;
    const float* pp = (lvl == 0) ? pr3 : ((lvl == 1) ? pr4 : pr5);
    mx = fmaxf(mx, pp[pi * 4 + j]);
  }
  red[t] = mx;
  __syncthreads();
  for (int s = 512; s > 0; s >>= 1) {
    if (t < s) red[t] = fmaxf(red[t], red[t + s]);
    __syncthreads();
  }
  float boxmax = red[0];
  __syncthreads();

  // ---- bitonic sort 4096 entries, descending by score ----
  for (int i = t; i < 4096; i += 1024) {
    key[i] = (i < NROI) ? scores[i] : -3.0e38f;
    idx[i] = i;
  }
  __syncthreads();
  for (int k = 2; k <= 4096; k <<= 1) {
    for (int j = k >> 1; j > 0; j >>= 1) {
      for (int i = t; i < 4096; i += 1024) {
        int ij = i ^ j;
        if (ij > i) {
          bool dir = ((i & k) == 0);  // true => descending block
          float ki = key[i], kj = key[ij];
          if ((ki < kj) == dir) {
            key[i] = kj; key[ij] = ki;
            int tmp = idx[i]; idx[i] = idx[ij]; idx[ij] = tmp;
          }
        }
      }
      __syncthreads();
    }
  }

  // ---- class-offset boxes, areas, initial keep ----
  for (int i = t; i < NROI; i += 1024) {
    int oi = idx[i];
    int lvl = oi / 1000, pi = oi % 1000;
    const float* pp = (lvl == 0) ? pr3 : ((lvl == 1) ? pr4 : pr5);
    float off = (float)classes[oi] * (boxmax + 1.0f);
    float x1 = pp[pi * 4 + 0] + off, y1 = pp[pi * 4 + 1] + off;
    float x2 = pp[pi * 4 + 2] + off, y2 = pp[pi * 4 + 3] + off;
    sb[i * 4 + 0] = x1; sb[i * 4 + 1] = y1;
    sb[i * 4 + 2] = x2; sb[i * 4 + 3] = y2;
    area[i] = fmaxf(x2 - x1, 0.0f) * fmaxf(y2 - y1, 0.0f);
    keep[i] = (key[i] >= 0.05f) ? 1 : 0;
  }
  __syncthreads();

  // ---- sequential greedy suppression (matches jax fori_loop exactly) ----
  for (int i = 0; i < NROI; ++i) {
    if (keep[i]) {   // uniform branch: all threads read same LDS word
      float x1i = sb[i * 4 + 0], y1i = sb[i * 4 + 1];
      float x2i = sb[i * 4 + 2], y2i = sb[i * 4 + 3], ai = area[i];
      for (int jj = i + 1 + t; jj < NROI; jj += 1024) {
        float lx = fmaxf(x1i, sb[jj * 4 + 0]);
        float ly = fmaxf(y1i, sb[jj * 4 + 1]);
        float rx = fminf(x2i, sb[jj * 4 + 2]);
        float ry = fminf(y2i, sb[jj * 4 + 3]);
        float w = fmaxf(rx - lx, 0.0f), h = fmaxf(ry - ly, 0.0f);
        float inter = w * h;
        float iou = inter / (ai + area[jj] - inter + 1e-8f);
        if (iou > 0.5f) keep[jj] = 0;
      }
    }
    __syncthreads();
  }

  for (int i = t; i < NROI; i += 1024)
    keep_out[idx[i]] = keep[i] ? 1.0f : 0.0f;
}

// =====================================================================
// Host-side orchestration
// =====================================================================
extern "C" void kernel_launch(void* const* d_in, const int* in_sizes, int n_in,
                              void* d_out, int out_size, void* d_ws, size_t ws_size,
                              hipStream_t stream) {
  const float* p3   = (const float*)d_in[0];
  const float* p4   = (const float*)d_in[1];
  const float* p5   = (const float*)d_in[2];
  const float* pr3  = (const float*)d_in[3];
  const float* pr4  = (const float*)d_in[4];
  const float* pr5  = (const float*)d_in[5];
  const float* w1   = (const float*)d_in[6];
  const float* b1   = (const float*)d_in[7];
  const float* w2   = (const float*)d_in[8];
  const float* b2   = (const float*)d_in[9];
  const float* wcls = (const float*)d_in[10];
  const float* bcls = (const float*)d_in[11];
  float* out = (float*)d_out;

  // ---- carve scratch (256B aligned) ----
  size_t off = 0;
  auto carve = [&](size_t bytes) {
    void* p = (char*)d_ws + off;
    off = (off + bytes + 255) & ~(size_t)255;
    return p;
  };
  _Float16* Amat  = (_Float16*)carve((size_t)MPAD * KFEAT * 2);   // 75.5 MB
  _Float16* W1p   = (_Float16*)carve((size_t)KFEAT * DHID * 2);   // 25.7 MB
  _Float16* h1    = (_Float16*)carve((size_t)MPAD * DHID * 2);    // 6.2 MB
  _Float16* W2p   = (_Float16*)carve((size_t)DHID * DHID * 2);    // 2.1 MB
  _Float16* h2    = (_Float16*)carve((size_t)MPAD * DHID * 2);    // 6.2 MB
  float*    logits  = (float*)carve((size_t)NROI * NCLS * 4);
  float*    scores  = (float*)carve((size_t)NROI * 4);
  int*      classes = (int*)carve((size_t)NROI * 4);
  (void)ws_size; (void)n_in; (void)in_sizes; (void)out_size;

  // 1) weight packing (fp32 -> fp16 WMMA B-fragment layout)
  {
    long tot = (long)KFEAT * DHID;
    pack_w_kernel<<<(int)((tot + 255) / 256), 256, 0, stream>>>(w1, W1p, KFEAT, DHID, DHID);
    tot = (long)DHID * DHID;
    pack_w_kernel<<<(int)((tot + 255) / 256), 256, 0, stream>>>(w2, W2p, DHID, DHID, DHID);
  }

  // 2) ROI-align -> fp16 A matrix (padded to 3008 rows)
  roi_align_kernel<<<MPAD, 256, 0, stream>>>(p3, p4, p5, pr3, pr4, pr5, Amat);

  // 3) GEMM1: (3008 x 12544) x (12544 x 1024) + b1, ReLU  -> h1 (fp16)
  //    188 m-tiles * 16 n-groups = 3008 wave-tasks; 376 blocks * 8 waves
  gemm_wmma_kernel<<<376, 256, 0, stream>>>(Amat, W1p, b1, h1, MT, DHID / 16, KFEAT, 1);

  // 4) GEMM2: (3008 x 1024) x (1024 x 1024) + b2, ReLU -> h2 (fp16)
  gemm_wmma_kernel<<<376, 256, 0, stream>>>(h1, W2p, b2, h2, MT, DHID / 16, DHID, 1);

  // 5) classifier logits (fp32 accumulate)
  classifier_kernel<<<(NROI * NCLS + 255) / 256, 256, 0, stream>>>(h2, wcls, bcls, logits);

  // 6) sigmoid / max / argmax, emit boxes+classes+scores
  score_kernel<<<(NROI + 255) / 256, 256, 0, stream>>>(logits, pr3, pr4, pr5, out, scores, classes);

  // 7) NMS -> keep mask
  size_t smem = (size_t)4096 * 4 + 4096 * 4 + NROI * 4 * 4 + NROI * 4 + NROI * 4 + 1024 * 4;
  nms_kernel<<<1, 1024, smem, stream>>>(pr3, pr4, pr5, scores, classes,
                                        out + NROI * 4 + NROI + NROI);
}
